// AttentionMechanism_41764261986823
// MI455X (gfx1250) — compile-verified, matrix operand order checked by reading, not run on hardware
//
#include <hip/hip_runtime.h>

typedef __attribute__((ext_vector_type(16))) _Float16 v16h;
typedef __attribute__((ext_vector_type(8)))  _Float16 v8h;
typedef __attribute__((ext_vector_type(8)))  float    v8f;
typedef unsigned int u32x4 __attribute__((ext_vector_type(4)));
typedef int          i32x4 __attribute__((ext_vector_type(4)));
typedef int          i32x8 __attribute__((ext_vector_type(8)));

#define S_TOT 4096
#define FEATC 128
#define EMBC  256
#define HEADS 8
#define HD    32
#define NQ    32
#define IQNN  64
#define ACT   8

// LDS B-tile row stride = K + 8 halves (16B pad per row -> conflict-free ds_load_b128)
#define BSTRIDE(K) ((K) + 8)

// ---------------------------------------------------------------------------
// Fragment loader (global): 16(row)x32(K) f16 tile, row-major, leading dim ld.
// ISA layout (CDNA5 ISA): lane l -> row r=l&15; half=l>>4:
//   VGPR0..3 : K = half*8 + 0..7        (16B contiguous)
//   VGPR4..7 : K = 16 + half*8 + 0..7   (16B contiguous)
// ---------------------------------------------------------------------------
__device__ inline v16h load_frag(const _Float16* __restrict__ base, int row0, int ld,
                                 int k0, int lane) {
    int r = lane & 15, half = lane >> 4;
    const _Float16* p = base + (size_t)(row0 + r) * ld + k0 + half * 8;
    v8h a0 = *reinterpret_cast<const v8h*>(p);
    v8h a1 = *reinterpret_cast<const v8h*>(p + 16);
    return __builtin_shufflevector(a0, a1, 0, 1, 2, 3, 4, 5, 6, 7,
                                           8, 9, 10, 11, 12, 13, 14, 15);
}

// ---------------------------------------------------------------------------
// GEMM: Out[M x 256] = epilogue(A[M x K] @ B'[256 x K]^T + bias)
// Block = 2 waves, each wave 16(M) x 64(N); grid = (M/32, 256/64).
// B tile (64 rows x K) staged into LDS once per block via the Tensor Data
// Mover (D# per CDNA5 ISA) when available.
// mode 0: leaky->f16 | 1: (x+b)*scale->f16 | 2: x+b->f16 | 3: leaky->f32
// ---------------------------------------------------------------------------
__global__ void gemm_wmma(const _Float16* __restrict__ A, const _Float16* __restrict__ B,
                          const float* __restrict__ bias, void* __restrict__ Out,
                          int K, int mode, float scale) {
    __shared__ __align__(16) _Float16 sB[64 * BSTRIDE(EMBC)];
    int lane = threadIdx.x & 31;
    int wave = threadIdx.x >> 5;
    int m0 = blockIdx.x * 32 + wave * 16;
    int n0 = blockIdx.y * 64;
    int strideH = BSTRIDE(K);

#if __has_builtin(__builtin_amdgcn_tensor_load_to_lds)
    if (wave == 0) {
        // D# group 0: count=1 | lds_addr | global_addr | type=2
        unsigned ldsBase = (unsigned)(uintptr_t)(void*)&sB[0];   // flat low 32 = LDS offset
        unsigned long long ga = (unsigned long long)(uintptr_t)(B + (size_t)n0 * K);
        u32x4 g0;
        g0[0] = 1u;                                              // count=1 (valid), user mode
        g0[1] = ldsBase;                                         // lds_addr (bytes)
        g0[2] = (unsigned)(ga & 0xFFFFFFFFu);                    // global_addr[31:0]
        g0[3] = (unsigned)((ga >> 32) & 0x01FFFFFFu) | (2u << 30); // addr[56:32] | type=2
        // D# group 1: data_size=2B, pad_enable, pad_interval = row DWORDs,
        // pad_amount = 4 DWORDs (16B); tensor = 64 x K, tile = 64 x K, stride0 = K
        int padI = (K == 256) ? 6 : 5;                           // 8<<6=512B / 8<<5=256B rows
        i32x8 g1;
        g1[0] = (1 << 16) | (1 << 20) | (padI << 22) | (3 << 25);
        g1[1] = (K & 0xFFFF) << 16;                              // tensor_dim0[15:0]
        g1[2] = (64 << 16);                                      // tensor_dim1[15:0]
        g1[3] = (K & 0xFFFF) << 16;                              // tile_dim0
        g1[4] = 64;                                              // tile_dim1
        g1[5] = K;                                               // tensor_dim0_stride[31:0]
        g1[6] = 0;
        g1[7] = 0;
        i32x4 z4a = {}, z4b = {};
#if defined(__clang_major__) && (__clang_major__ >= 23)
        i32x8 z8 = {};
        __builtin_amdgcn_tensor_load_to_lds(g0, g1, z4a, z4b, z8, 0);
#else
        __builtin_amdgcn_tensor_load_to_lds(g0, g1, z4a, z4b, 0);
#endif
        __builtin_amdgcn_s_wait_tensorcnt(0);
    }
    __syncthreads();
#else
    // Fallback: cooperative copy of the 64 x K B-slab into padded LDS
    for (int i = threadIdx.x; i < 64 * (K / 8); i += 64) {
        int r = i / (K / 8), kk = (i % (K / 8)) * 8;
        *reinterpret_cast<v8h*>(&sB[(size_t)r * strideH + kk]) =
            *reinterpret_cast<const v8h*>(&B[(size_t)(n0 + r) * K + kk]);
    }
    __syncthreads();
#endif

    v8f acc[4] = {};
    int r = lane & 15, half = lane >> 4;
    for (int k0 = 0; k0 < K; k0 += 32) {
        v16h a = load_frag(A, m0, K, k0, lane);
#pragma unroll
        for (int t = 0; t < 4; ++t) {
            const _Float16* p = &sB[(size_t)(t * 16 + r) * strideH + k0 + half * 8];
            v8h b0 = *reinterpret_cast<const v8h*>(p);
            v8h b1 = *reinterpret_cast<const v8h*>(p + 16);
            v16h b = __builtin_shufflevector(b0, b1, 0, 1, 2, 3, 4, 5, 6, 7,
                                                     8, 9, 10, 11, 12, 13, 14, 15);
            acc[t] = __builtin_amdgcn_wmma_f32_16x16x32_f16(
                false, a, false, b, (short)0, acc[t], false, false);
        }
    }
    // C/D layout: lane l -> column n = l&15, rows m = (l>>4)*8 + vgpr
#pragma unroll
    for (int t = 0; t < 4; ++t) {
        int n = n0 + t * 16 + r;
        float bv = bias[n];
#pragma unroll
        for (int j = 0; j < 8; ++j) {
            int m = m0 + half * 8 + j;
            float v = acc[t][j] + bv;
            if (mode == 1) v *= scale;
            if (mode == 0 || mode == 3) v = v >= 0.f ? v : 0.01f * v;
            if (mode == 3) ((float*)Out)[(size_t)m * EMBC + n] = v;
            else           ((_Float16*)Out)[(size_t)m * EMBC + n] = (_Float16)v;
        }
    }
}

// ---------------------------------------------------------------------------
// x = leaky(world_state + sinusoidal PE), store f16 [s][c] with s = w*64 + h
// ---------------------------------------------------------------------------
__global__ void prep_x(const float* __restrict__ ws, _Float16* __restrict__ X) {
    int s = blockIdx.x;
    int c = threadIdx.x;           // 0..127
    int w = s >> 6, h = s & 63;
    int cc = (c < 64) ? c : (c - 64);
    int i = cc >> 1;
    float div = __expf(-(float)i * (2.0f * 9.210340371976184f / 64.0f));
    float arg = (float)((c < 64) ? h : w) * div;
    float pe = (c & 1) ? cosf(arg) : sinf(arg);
    float v = ws[(size_t)c * S_TOT + s] + pe;
    v = v >= 0.f ? v : 0.01f * v;
    X[(size_t)s * FEATC + c] = (_Float16)v;
}

__global__ void cvt16(const float* __restrict__ in, _Float16* __restrict__ out, int n) {
    int i = blockIdx.x * 256 + threadIdx.x;
    if (i < n) out[i] = (_Float16)in[i];
}

// ---------------------------------------------------------------------------
// IQN: qnf[n][o] = leaky( sum_j cos((j+1)*pi*tau[n]) * w_iqn[o][j] + b_iqn[o] )
// Also writes tau into the tail of d_out.
// ---------------------------------------------------------------------------
__global__ void iqn_kernel(const float* __restrict__ tau, const float* __restrict__ w_iqn,
                           const float* __restrict__ b_iqn, float* __restrict__ qnf,
                           float* __restrict__ out_tail) {
    int o = threadIdx.x;           // 0..255
    __shared__ float cosv[NQ][IQNN];
    for (int idx = o; idx < NQ * IQNN; idx += 256) {
        int n = idx >> 6, j = idx & 63;
        cosv[n][j] = cosf((float)(j + 1) * 3.14159265358979323846f * tau[n]);
    }
    __syncthreads();
    for (int n = 0; n < NQ; ++n) {
        float a = b_iqn[o];
        for (int j = 0; j < IQNN; ++j) a += cosv[n][j] * w_iqn[o * IQNN + j];
        a = a >= 0.f ? a : 0.01f * a;
        qnf[n * EMBC + o] = a;
    }
    if (o < NQ) out_tail[o] = tau[o];
}

// ---------------------------------------------------------------------------
// Windowed attention (Chebyshev radius 4 on (w,h)); online softmax.
// Block = 256 threads = 8 waves = 8 heads for one query s; lane = head dim.
// ---------------------------------------------------------------------------
__global__ void attn_win(const _Float16* __restrict__ Q, const _Float16* __restrict__ Km,
                         const _Float16* __restrict__ Vm, _Float16* __restrict__ AO) {
    int s = blockIdx.x;
    int head = threadIdx.x >> 5;
    int lane = threadIdx.x & 31;
    int w = s >> 6, h = s & 63;
    int col = head * HD + lane;
    float q = (float)Q[(size_t)s * EMBC + col];
    float mmax = -3.4e38f, denom = 0.f, acc = 0.f;
    for (int dw = -4; dw <= 4; ++dw) {
        int wq = w + dw;
        if ((unsigned)wq >= 64u) continue;
        // prefetch the next row's K/V cachelines while this row computes
        int wn = wq + 1;
        if (wn <= w + 4 && wn < 64) {
            __builtin_prefetch(&Km[(size_t)(wn * 64 + h) * EMBC + col], 0, 1);
            __builtin_prefetch(&Vm[(size_t)(wn * 64 + h) * EMBC + col], 0, 1);
        }
        for (int dh = -4; dh <= 4; ++dh) {
            int hh = h + dh;
            if ((unsigned)hh >= 64u) continue;
            int t = wq * 64 + hh;
            float x = q * (float)Km[(size_t)t * EMBC + col];
            for (int off = 16; off > 0; off >>= 1) x += __shfl_xor(x, off, 32);
            float vv = (float)Vm[(size_t)t * EMBC + col];
            if (x > mmax) {
                float sc = __expf(mmax - x);
                denom *= sc; acc *= sc; mmax = x;
            }
            float p = __expf(x - mmax);
            denom += p; acc += p * vv;
        }
    }
    AO[(size_t)s * EMBC + col] = (_Float16)(acc / denom);
}

// ---------------------------------------------------------------------------
// Per-quantile heads: out[n,e,w,h] = attn[s][e]*qnf[n][e];
// A = leaky(out @ w_a^T + b_a); V = leaky(out @ w_vh^T + b_vh);
// Q = V + A - mean_h(A).   grid = (w, n), block = 64 (h).
// ---------------------------------------------------------------------------
__global__ void head_kernel(const float* __restrict__ attnf, const float* __restrict__ qnf,
                            const float* __restrict__ w_a, const float* __restrict__ b_a,
                            const float* __restrict__ w_vh, const float* __restrict__ b_vh,
                            float* __restrict__ out) {
    int w = blockIdx.x, n = blockIdx.y, h = threadIdx.x;
    __shared__ float sWa[ACT][EMBC];
    __shared__ float sWv[EMBC];
    __shared__ float sQn[EMBC];
    for (int i = h; i < ACT * EMBC; i += 64) sWa[i >> 8][i & 255] = w_a[i];
    for (int i = h; i < EMBC; i += 64) { sWv[i] = w_vh[i]; sQn[i] = qnf[n * EMBC + i]; }
    __syncthreads();
    int s = w * 64 + h;
    float accA[ACT];
    float accV = b_vh[0];
#pragma unroll
    for (int a = 0; a < ACT; ++a) accA[a] = b_a[a];
    for (int e = 0; e < EMBC; ++e) {
        float val = attnf[(size_t)s * EMBC + e] * sQn[e];
        accV += val * sWv[e];
#pragma unroll
        for (int a = 0; a < ACT; ++a) accA[a] += val * sWa[a][e];
    }
    float V = accV >= 0.f ? accV : 0.01f * accV;
    __shared__ float red[ACT][64];
    __shared__ float mean[ACT];
#pragma unroll
    for (int a = 0; a < ACT; ++a) {
        float Av = accA[a] >= 0.f ? accA[a] : 0.01f * accA[a];
        accA[a] = Av;
        red[a][h] = Av;
    }
    __syncthreads();
    if (h < ACT) {
        float sum = 0.f;
        for (int i = 0; i < 64; ++i) sum += red[h][i];
        mean[h] = sum * (1.0f / 64.0f);
    }
    __syncthreads();
#pragma unroll
    for (int a = 0; a < ACT; ++a)
        out[(((size_t)n * ACT + a) * 64 + w) * 64 + h] = V + accA[a] - mean[a];
}

// ---------------------------------------------------------------------------
extern "C" void kernel_launch(void* const* d_in, const int* in_sizes, int n_in,
                              void* d_out, int out_size, void* d_ws, size_t ws_size,
                              hipStream_t stream) {
    const float* ws    = (const float*)d_in[0];
    const float* tau   = (const float*)d_in[1];
    const float* wq_c  = (const float*)d_in[2];
    const float* bq_c  = (const float*)d_in[3];
    const float* wk_c  = (const float*)d_in[4];
    const float* bk_c  = (const float*)d_in[5];
    const float* wv_c  = (const float*)d_in[6];
    const float* bv_c  = (const float*)d_in[7];
    const float* w_q   = (const float*)d_in[8];
    const float* b_q   = (const float*)d_in[9];
    const float* w_k   = (const float*)d_in[10];
    const float* b_k   = (const float*)d_in[11];
    const float* w_v   = (const float*)d_in[12];
    const float* b_v   = (const float*)d_in[13];
    const float* w_o   = (const float*)d_in[14];
    const float* b_o   = (const float*)d_in[15];
    const float* w_a   = (const float*)d_in[16];
    const float* b_a   = (const float*)d_in[17];
    const float* w_vh  = (const float*)d_in[18];
    const float* b_vh  = (const float*)d_in[19];
    const float* w_iqn = (const float*)d_in[20];
    const float* b_iqn = (const float*)d_in[21];
    (void)in_sizes; (void)n_in; (void)out_size; (void)ws_size;

    char* base = (char*)d_ws;
    size_t off = 0;
    auto alloc = [&](size_t bytes) -> void* {
        void* p = base + off;
        off = (off + bytes + 255) & ~(size_t)255;
        return p;
    };
    _Float16* X16   = (_Float16*)alloc((size_t)S_TOT * FEATC * 2);
    _Float16* Wqc16 = (_Float16*)alloc((size_t)EMBC * FEATC * 2);
    _Float16* Wkc16 = (_Float16*)alloc((size_t)EMBC * FEATC * 2);
    _Float16* Wvc16 = (_Float16*)alloc((size_t)EMBC * FEATC * 2);
    _Float16* Wq16  = (_Float16*)alloc((size_t)EMBC * EMBC * 2);
    _Float16* Wk16  = (_Float16*)alloc((size_t)EMBC * EMBC * 2);
    _Float16* Wv16  = (_Float16*)alloc((size_t)EMBC * EMBC * 2);
    _Float16* Wo16  = (_Float16*)alloc((size_t)EMBC * EMBC * 2);
    _Float16* Qin16 = (_Float16*)alloc((size_t)S_TOT * EMBC * 2);
    _Float16* Kin16 = (_Float16*)alloc((size_t)S_TOT * EMBC * 2);
    _Float16* Vin16 = (_Float16*)alloc((size_t)S_TOT * EMBC * 2);
    _Float16* Q16   = (_Float16*)alloc((size_t)S_TOT * EMBC * 2);
    _Float16* K16   = (_Float16*)alloc((size_t)S_TOT * EMBC * 2);
    _Float16* V16   = (_Float16*)alloc((size_t)S_TOT * EMBC * 2);
    _Float16* AO16  = (_Float16*)alloc((size_t)S_TOT * EMBC * 2);
    float*    ATTNF = (float*)alloc((size_t)S_TOT * EMBC * 4);
    float*    QNF   = (float*)alloc((size_t)NQ * EMBC * 4);
    float*    outF  = (float*)d_out;

    // 1) weights -> f16
    cvt16<<<(EMBC * FEATC + 255) / 256, 256, 0, stream>>>(wq_c, Wqc16, EMBC * FEATC);
    cvt16<<<(EMBC * FEATC + 255) / 256, 256, 0, stream>>>(wk_c, Wkc16, EMBC * FEATC);
    cvt16<<<(EMBC * FEATC + 255) / 256, 256, 0, stream>>>(wv_c, Wvc16, EMBC * FEATC);
    cvt16<<<(EMBC * EMBC + 255) / 256, 256, 0, stream>>>(w_q, Wq16, EMBC * EMBC);
    cvt16<<<(EMBC * EMBC + 255) / 256, 256, 0, stream>>>(w_k, Wk16, EMBC * EMBC);
    cvt16<<<(EMBC * EMBC + 255) / 256, 256, 0, stream>>>(w_v, Wv16, EMBC * EMBC);
    cvt16<<<(EMBC * EMBC + 255) / 256, 256, 0, stream>>>(w_o, Wo16, EMBC * EMBC);

    // 2) x = leaky(ws + PE)
    prep_x<<<S_TOT, FEATC, 0, stream>>>(ws, X16);

    // 3) IQN embedding + tau tail of output
    iqn_kernel<<<1, 256, 0, stream>>>(tau, w_iqn, b_iqn, QNF, outF + (size_t)NQ * ACT * S_TOT);

    // 4) conv1x1 GEMMs (K=128) with leaky -> f16
    dim3 gdim(S_TOT / 32, EMBC / 64);
    gemm_wmma<<<gdim, 64, 0, stream>>>(X16, Wqc16, bq_c, Qin16, FEATC, 0, 1.f);
    gemm_wmma<<<gdim, 64, 0, stream>>>(X16, Wkc16, bk_c, Kin16, FEATC, 0, 1.f);
    gemm_wmma<<<gdim, 64, 0, stream>>>(X16, Wvc16, bv_c, Vin16, FEATC, 0, 1.f);

    // 5) projection GEMMs (K=256); Q scaled by hd^-0.5 = 1/sqrt(32)
    gemm_wmma<<<gdim, 64, 0, stream>>>(Qin16, Wq16, b_q, Q16, EMBC, 1, 0.17677669529663687f);
    gemm_wmma<<<gdim, 64, 0, stream>>>(Kin16, Wk16, b_k, K16, EMBC, 2, 1.f);
    gemm_wmma<<<gdim, 64, 0, stream>>>(Vin16, Wv16, b_v, V16, EMBC, 2, 1.f);

    // 6) windowed attention (9x9 Chebyshev window), online softmax
    attn_win<<<S_TOT, 256, 0, stream>>>(Q16, K16, V16, AO16);

    // 7) output projection + leaky -> f32
    gemm_wmma<<<gdim, 64, 0, stream>>>(AO16, Wo16, b_o, ATTNF, EMBC, 3, 1.f);

    // 8) quantile scaling + A/V heads + dueling combine
    head_kernel<<<dim3(64, NQ), 64, 0, stream>>>(ATTNF, QNF, w_a, b_a, w_vh, b_vh, outF);
}
